// SemanticSpaceInformedPrompting_88914412962027
// MI455X (gfx1250) — matrix operand order repaired and checked by reading, not compiled, not run on hardware
//
#include <hip/hip_runtime.h>
#include <hip/hip_bf16.h>

typedef __attribute__((ext_vector_type(16))) _Float16 v16h;
typedef __attribute__((ext_vector_type(8)))  _Float16 v8h;
typedef __attribute__((ext_vector_type(8)))  float    v8f;

#define BATCH  64
#define SEQ    512
#define DIM    256
#define NANCH  200000
#define TOPK   16
#define NEG_INF (-3.0e38f)

// ---------------------------------------------------------------------------
// Kernel 1: query = mean(x, axis=1); q_norm = query / max(||query||, eps)
// stored as f16 for the WMMA A-matrix. One block per batch, thread = dim.
// ---------------------------------------------------------------------------
__global__ __launch_bounds__(256)
void query_kernel(const float* __restrict__ x, _Float16* __restrict__ qn)
{
    const int b = blockIdx.x;
    const int d = threadIdx.x;
    const float* px = x + (size_t)b * SEQ * DIM + d;
    float s = 0.f;
    #pragma unroll 8
    for (int l = 0; l < SEQ; ++l) s += px[(size_t)l * DIM];
    const float mean = s * (1.0f / SEQ);

    __shared__ float red[256];
    red[d] = mean * mean;
    __syncthreads();
    #pragma unroll
    for (int off = 128; off > 0; off >>= 1) {
        if (d < off) red[d] += red[d + off];
        __syncthreads();
    }
    __shared__ float invs;
    if (d == 0) invs = 1.0f / fmaxf(sqrtf(red[0]), 1e-12f);
    __syncthreads();
    qn[b * DIM + d] = (_Float16)(mean * invs);
}

// ---------------------------------------------------------------------------
// Kernel 2: inv_norm[n] = 1 / max(||anchors[n]||, eps).  One wave32 per row:
// lane loads 2 float4 (8 floats of 256), shfl_xor reduction.
// ---------------------------------------------------------------------------
__global__ __launch_bounds__(256)
void anorm_kernel(const float* __restrict__ anchors, float* __restrict__ inv_norm)
{
    const int lane = threadIdx.x & 31;
    const int row  = blockIdx.x * 8 + (threadIdx.x >> 5);
    if (row >= NANCH) return;
    const float4* p = (const float4*)(anchors + (size_t)row * DIM);
    float4 v = p[lane];
    float ss = v.x * v.x + v.y * v.y + v.z * v.z + v.w * v.w;
    v = p[lane + 32];
    ss += v.x * v.x + v.y * v.y + v.z * v.z + v.w * v.w;
    #pragma unroll
    for (int m = 16; m > 0; m >>= 1) ss += __shfl_xor(ss, m, 32);
    if (lane == 0) inv_norm[row] = 1.0f / fmaxf(sqrtf(ss), 1e-12f);
}

// ---------------------------------------------------------------------------
// Kernel 3: similarity GEMM via v_wmma_f32_16x16x32_f16.
// S[b, n] = inv_norm[n] * sum_d qn[b,d] * anchors[n,d]   (stored as f16)
// Block = 256 threads (8 waves), covers 256 anchors (wave = 2 N-tiles of 16).
// A (q_norm f16, 64x256 = 32KB) staged in LDS once per block.
// K = 256 -> 8 WMMA steps of 32.  Accumulators: acc[ntile][mtile] (4 M-tiles).
// ---------------------------------------------------------------------------
__global__ __launch_bounds__(256)
void sim_wmma_kernel(const float*    __restrict__ anchors,
                     const _Float16* __restrict__ qn,
                     const float*    __restrict__ inv_norm,
                     _Float16*       __restrict__ sims)
{
    __shared__ _Float16 As[BATCH * DIM];   // 32 KB

    const int tid = threadIdx.x;
    {   // cooperative 128B-per-thread LDS fill of the A matrix
        const uint4* src = (const uint4*)qn;
        uint4* dst = (uint4*)As;
        #pragma unroll
        for (int i = 0; i < (BATCH * DIM * 2 / 16) / 256; ++i)
            dst[tid + i * 256] = src[tid + i * 256];
    }
    __syncthreads();

    const int wave  = tid >> 5;
    const int lane  = tid & 31;
    const int part  = lane >> 4;    // 0: lanes 0-15, 1: lanes 16-31
    const int nl    = lane & 15;
    const int nbase = blockIdx.x * 256 + wave * 32;

    v8f acc[2][4] = {};

    #pragma unroll
    for (int ks = 0; ks < 8; ++ks) {
        const int k0 = ks * 32;

        // B fragments: 32(K) x 16(N) f16.  Lane nl holds anchor (nbase+nt*16+nl),
        // lanes 0-15 carry K=k0..k0+15, lanes 16-31 carry K=k0+16..k0+31.
        v16h bfrag[2];
        #pragma unroll
        for (int nt = 0; nt < 2; ++nt) {
            int r = nbase + nt * 16 + nl;
            if (r >= NANCH) r = NANCH - 1;                 // clamp (guard on store)
            const float* bp = anchors + (size_t)r * DIM + k0 + part * 16;
            __builtin_prefetch(bp + 32, 0, 1);             // next K slab -> global_prefetch
            const float4 f0 = *(const float4*)(bp + 0);
            const float4 f1 = *(const float4*)(bp + 4);
            const float4 f2 = *(const float4*)(bp + 8);
            const float4 f3 = *(const float4*)(bp + 12);
            v16h bf;
            bf[0]  = (_Float16)f0.x; bf[1]  = (_Float16)f0.y;
            bf[2]  = (_Float16)f0.z; bf[3]  = (_Float16)f0.w;
            bf[4]  = (_Float16)f1.x; bf[5]  = (_Float16)f1.y;
            bf[6]  = (_Float16)f1.z; bf[7]  = (_Float16)f1.w;
            bf[8]  = (_Float16)f2.x; bf[9]  = (_Float16)f2.y;
            bf[10] = (_Float16)f2.z; bf[11] = (_Float16)f2.w;
            bf[12] = (_Float16)f3.x; bf[13] = (_Float16)f3.y;
            bf[14] = (_Float16)f3.z; bf[15] = (_Float16)f3.w;
            bfrag[nt] = bf;
        }

        // A fragments: 16(M) x 32(K) f16 from LDS, contiguous k order (same as B):
        //   lane (part=0): V0..7 K=k0+0..15
        //   lane (part=1): V0..7 K=k0+16..31
        #pragma unroll
        for (int mt = 0; mt < 4; ++mt) {
            const _Float16* ap = As + (size_t)(mt * 16 + nl) * DIM + k0 + part * 16;
            const v8h a0 = *(const v8h*)(ap);
            const v8h a1 = *(const v8h*)(ap + 8);
            const v16h af = __builtin_shufflevector(a0, a1,
                0, 1, 2, 3, 4, 5, 6, 7, 8, 9, 10, 11, 12, 13, 14, 15);
            acc[0][mt] = __builtin_amdgcn_wmma_f32_16x16x32_f16(
                false, af, false, bfrag[0], (short)0, acc[0][mt], false, false);
            acc[1][mt] = __builtin_amdgcn_wmma_f32_16x16x32_f16(
                false, af, false, bfrag[1], (short)0, acc[1][mt], false, false);
        }
    }

    // Epilogue: scale by inv_norm[n], store f16 sims.
    // C layout: VGPR j, lanes 0-15 -> M=j ; lanes 16-31 -> M=j+8 ; N = lane&15.
    #pragma unroll
    for (int nt = 0; nt < 2; ++nt) {
        const int n = nbase + nt * 16 + nl;
        if (n >= NANCH) continue;
        const float inv = inv_norm[n];
        #pragma unroll
        for (int mt = 0; mt < 4; ++mt) {
            #pragma unroll
            for (int j = 0; j < 8; ++j) {
                const int b = mt * 16 + part * 8 + j;
                sims[(size_t)b * NANCH + n] = (_Float16)(acc[nt][mt][j] * inv);
            }
        }
    }
}

// ---------------------------------------------------------------------------
// Kernel 4: per-batch top-16.  One block per batch row.  Each thread keeps a
// sorted 16-entry list in LDS over its strided slice, then 16 rounds of
// block-wide argmax merge produce descending top-k indices.
// ---------------------------------------------------------------------------
__global__ __launch_bounds__(256)
void topk_kernel(const _Float16* __restrict__ sims, int* __restrict__ topidx)
{
    const int b   = blockIdx.x;
    const int tid = threadIdx.x;

    __shared__ float lv[256][TOPK];   // 16 KB
    __shared__ int   li[256][TOPK];   // 16 KB
    __shared__ int   hp[256];
    __shared__ float redv[256];
    __shared__ int   redt[256];

    #pragma unroll
    for (int k = 0; k < TOPK; ++k) { lv[tid][k] = NEG_INF; li[tid][k] = 0; }
    hp[tid] = 0;

    const _Float16* row = sims + (size_t)b * NANCH;
    for (int i = tid; i < NANCH; i += 256) {
        const float v = (float)row[i];
        if (v > lv[tid][TOPK - 1]) {
            int p = TOPK - 1;
            while (p > 0 && lv[tid][p - 1] < v) {
                lv[tid][p] = lv[tid][p - 1];
                li[tid][p] = li[tid][p - 1];
                --p;
            }
            lv[tid][p] = v;
            li[tid][p] = i;
        }
    }
    __syncthreads();

    for (int k = 0; k < TOPK; ++k) {
        const int h = hp[tid];
        redv[tid] = (h < TOPK) ? lv[tid][h] : NEG_INF;
        redt[tid] = tid;
        __syncthreads();
        #pragma unroll
        for (int s = 128; s > 0; s >>= 1) {
            if (tid < s && redv[tid + s] > redv[tid]) {
                redv[tid] = redv[tid + s];
                redt[tid] = redt[tid + s];
            }
            __syncthreads();
        }
        if (tid == 0) {
            const int w = redt[0];
            topidx[b * TOPK + k] = li[w][hp[w]];
            hp[w]++;
        }
        __syncthreads();
    }
}

// ---------------------------------------------------------------------------
// Kernel 5: out[b] = concat(anchors[top_idx[b]], x[b]) along seq axis.
// Vectorized float4 streaming copy / gather.
// ---------------------------------------------------------------------------
__global__ __launch_bounds__(256)
void gather_kernel(const float* __restrict__ x,
                   const float* __restrict__ anchors,
                   const int*   __restrict__ topidx,
                   float*       __restrict__ out)
{
    const long long total4 = (long long)BATCH * (TOPK + SEQ) * DIM / 4;
    for (long long t = (long long)blockIdx.x * 256 + threadIdx.x;
         t < total4; t += (long long)gridDim.x * 256) {
        const long long e = t * 4;
        const int d        = (int)(e & (DIM - 1));
        const long long rj = e / DIM;
        const int j        = (int)(rj % (TOPK + SEQ));
        const int b        = (int)(rj / (TOPK + SEQ));
        const float4* src;
        if (j < TOPK) {
            const int a = topidx[b * TOPK + j];
            src = (const float4*)(anchors + (size_t)a * DIM + d);
        } else {
            src = (const float4*)(x + ((size_t)b * SEQ + (j - TOPK)) * DIM + d);
        }
        ((float4*)out)[t] = *src;
    }
}

// ---------------------------------------------------------------------------
// Host-side launcher.  Workspace layout (26.44 MB total):
//   [0)            q_norm f16       32768 B
//   [32768)        inv_norm f32    800000 B
//   [832768)       top_idx  i32      4096 B
//   [836864)       sims     f16  25600000 B
// top_k is known statically (=16) from the reference; d_in[2] is a device
// scalar and cannot be read during graph capture.
// ---------------------------------------------------------------------------
extern "C" void kernel_launch(void* const* d_in, const int* in_sizes, int n_in,
                              void* d_out, int out_size, void* d_ws, size_t ws_size,
                              hipStream_t stream)
{
    (void)in_sizes; (void)n_in; (void)out_size; (void)ws_size;

    const float* x       = (const float*)d_in[0];
    const float* anchors = (const float*)d_in[1];
    float* out = (float*)d_out;

    char* ws = (char*)d_ws;
    _Float16* qn       = (_Float16*)(ws);
    float*    inv_norm = (float*)   (ws + 32768);
    int*      topidx   = (int*)     (ws + 832768);
    _Float16* sims     = (_Float16*)(ws + 836864);

    query_kernel   <<<BATCH, 256, 0, stream>>>(x, qn);
    anorm_kernel   <<<(NANCH + 7) / 8, 256, 0, stream>>>(anchors, inv_norm);
    sim_wmma_kernel<<<(NANCH + 255) / 256, 256, 0, stream>>>(anchors, qn, inv_norm, sims);
    topk_kernel    <<<BATCH, 256, 0, stream>>>(sims, topidx);
    gather_kernel  <<<8448, 256, 0, stream>>>(x, anchors, topidx, out);
}
